// Context2Query_9320079032491
// MI455X (gfx1250) — compile-verified, matrix operand order checked by reading, not run on hardware
//
#include <hip/hip_runtime.h>
#include <hip/hip_bf16.h>
#include <math.h>

// out[b,d,n] = sum_m softmax(s[b,n,:])[m] * u[b,m,d]   B=16,N=2048,M=2048,D=1024
//
// 1) softmax_rows : s[B,N,M] f32 -> P[B,N,M] bf16            (128 MiB ws)
// 2) transpose_u  : u[B,M,D] f32 -> Ut[B,D,M] bf16           ( 64 MiB ws)
// 3) c2q_gemm     : v_wmma_f32_16x16x32_bf16, K-step 64, double-buffered LDS
//                   filled with global_load_async_to_lds_b128 (ASYNCcnt DMA).
// Requires ws_size >= 192 MiB.  P+Ut (~192 MB) is L2-resident on MI455X.

typedef __attribute__((ext_vector_type(16))) __bf16       v16bf;
typedef __attribute__((ext_vector_type(8)))  float        v8f;
typedef __attribute__((ext_vector_type(4)))  unsigned int v4u;

union FragU { v16bf v; v4u q[2]; };

static constexpr int Bc = 16;
static constexpr int Nc = 2048;
static constexpr int Mc = 2048;
static constexpr int Dc = 1024;

static constexpr int LDS_STRIDE = 72;   // bf16 elems/row: 144 B, 16B-aligned,
                                        // r*36 mod 64 banks = perfect permutation

static __device__ __forceinline__ unsigned short f2bf(float f) {
  unsigned int x = __float_as_uint(f);
  x += 0x7fffu + ((x >> 16) & 1u);      // round-to-nearest-even
  return (unsigned short)(x >> 16);
}

// ---------------------------------------------------------------------------
// Kernel 1: row softmax, f32 -> bf16 probabilities. One block per row.
// ---------------------------------------------------------------------------
__global__ __launch_bounds__(256) void softmax_rows(const float* __restrict__ s,
                                                    unsigned short* __restrict__ p) {
  __shared__ float red[256];
  const size_t row = blockIdx.x;
  const float* sr = s + row * (size_t)Mc;
  unsigned short* pr = p + row * (size_t)Mc;
  const int t = threadIdx.x;

  float v[8];
  float mx = -INFINITY;
#pragma unroll
  for (int k = 0; k < 8; ++k) {
    v[k] = sr[t + 256 * k];
    mx = fmaxf(mx, v[k]);
  }
  red[t] = mx;
  __syncthreads();
  for (int off = 128; off > 0; off >>= 1) {
    if (t < off) red[t] = fmaxf(red[t], red[t + off]);
    __syncthreads();
  }
  mx = red[0];
  __syncthreads();

  float sum = 0.0f;
#pragma unroll
  for (int k = 0; k < 8; ++k) {
    v[k] = __expf(v[k] - mx);
    sum += v[k];
  }
  red[t] = sum;
  __syncthreads();
  for (int off = 128; off > 0; off >>= 1) {
    if (t < off) red[t] += red[t + off];
    __syncthreads();
  }
  const float inv = 1.0f / red[0];

#pragma unroll
  for (int k = 0; k < 8; ++k)
    pr[t + 256 * k] = f2bf(v[k] * inv);
}

// ---------------------------------------------------------------------------
// Kernel 2: u[B,M,D] f32 -> Ut[B,D,M] bf16 via 32x32 LDS tiles (both sides
// coalesced).
// ---------------------------------------------------------------------------
__global__ __launch_bounds__(256) void transpose_u(const float* __restrict__ u,
                                                   unsigned short* __restrict__ ut) {
  __shared__ unsigned short tile[32][33];
  const int b  = blockIdx.z;
  const int m0 = blockIdx.x * 32;
  const int d0 = blockIdx.y * 32;
  const int tx = threadIdx.x & 31;
  const int ty = threadIdx.x >> 5;

  const float* ub = u + ((size_t)b * Mc + m0) * Dc + d0;
#pragma unroll
  for (int r = 0; r < 4; ++r) {
    int i = ty + 8 * r;
    tile[i][tx] = f2bf(ub[(size_t)i * Dc + tx]);
  }
  __syncthreads();
  unsigned short* ob = ut + ((size_t)b * Dc + d0) * Mc + m0;
#pragma unroll
  for (int r = 0; r < 4; ++r) {
    int j = ty + 8 * r;
    ob[(size_t)j * Mc + tx] = tile[tx][j];
  }
}

// ---------------------------------------------------------------------------
// Kernel 3: WMMA GEMM.  A = Ut tile (rows=D, K=M), B = P tile (cols=N, K=M).
// 256 threads = 8 wave32; 128(D) x 128(N) block tile; K-step 64.
// Double-buffered LDS filled by global_load_async_to_lds_b128:
//   - each thread issues 8 async b128 loads per K-step (4 per operand tile)
//   - async completions are in-order per wave, so after issuing the next
//     batch of 8, s_wait_asynccnt <= 8 guarantees the *current* buffer landed.
// ---------------------------------------------------------------------------
__global__ __launch_bounds__(256) void c2q_gemm(const unsigned short* __restrict__ P,
                                                const unsigned short* __restrict__ Ut,
                                                float* __restrict__ out) {
  __shared__ unsigned short lA[2][128 * LDS_STRIDE];   // Ut tile [d][k]
  __shared__ unsigned short lB[2][128 * LDS_STRIDE];   // P  tile [n][k]

  const int b  = blockIdx.z;
  const int n0 = blockIdx.x * 128;
  const int d0 = blockIdx.y * 128;
  const int t    = threadIdx.x;
  const int lane = t & 31;
  const int wave = t >> 5;
  const int dw = (wave >> 1) * 32;
  const int nw = (wave & 1) * 64;

  const unsigned short* gA = Ut + ((size_t)b * Dc + d0) * Mc;
  const unsigned short* gB = P  + ((size_t)b * Nc + n0) * Mc;

  // async staging map: chunk c = t&7 (16 B within the 128 B K-row),
  // rows r0 + 32q, q = 0..3 -> 4 b128 loads per tile per thread.
  const int ce = (t & 7) * 8;            // K element offset of this thread's chunk
  const int r0 = t >> 3;                 // 0..31

  v8f acc[2][4];
#pragma unroll
  for (int i = 0; i < 2; ++i)
#pragma unroll
    for (int j = 0; j < 4; ++j)
      acc[i][j] = (v8f){0.f, 0.f, 0.f, 0.f, 0.f, 0.f, 0.f, 0.f};

  const int kh = (lane >> 4) * 8;        // half-wave K chunk select (ISA layout)
  const int ra = dw + (lane & 15);
  const int rb = nw + (lane & 15);

  auto issue_tiles = [&](int buf, int k0) {
#pragma unroll
    for (int q = 0; q < 4; ++q) {
      const int r = r0 + 32 * q;
      const unsigned goff = (unsigned)(((r * Mc) + k0 + ce) * 2);   // bytes
      const unsigned la = (unsigned)(uintptr_t)(&lA[buf][r * LDS_STRIDE + ce]);
      const unsigned lb = (unsigned)(uintptr_t)(&lB[buf][r * LDS_STRIDE + ce]);
      asm volatile("global_load_async_to_lds_b128 %0, %1, %2"
                   :: "v"(la), "v"(goff), "s"(gA) : "memory");
      asm volatile("global_load_async_to_lds_b128 %0, %1, %2"
                   :: "v"(lb), "v"(goff), "s"(gB) : "memory");
    }
  };

  issue_tiles(0, 0);                     // prologue: fill buffer 0

  for (int it = 0; it < Mc / 64; ++it) {
    const int cur = it & 1;
    const int k1 = (it + 1) * 64;

    if (k1 < Mc) {
      issue_tiles(cur ^ 1, k1);          // DMA next tile during this compute
      asm volatile("s_wait_asynccnt 0x8" ::: "memory");   // current buf landed
    } else {
      asm volatile("s_wait_asynccnt 0x0" ::: "memory");
    }
    __syncthreads();                     // all waves' DMA for cur visible

    const unsigned short* sA = lA[cur];
    const unsigned short* sB = lB[cur];
#pragma unroll
    for (int w = 0; w < 2; ++w) {        // two 32-K sub-steps of the 64-K tile
      FragU af[2], bfr[4];
#pragma unroll
      for (int mt = 0; mt < 2; ++mt) {
        const unsigned short* pa = sA + (ra + mt * 16) * LDS_STRIDE + w * 32 + kh;
        af[mt].q[0] = *(const v4u*)(pa);
        af[mt].q[1] = *(const v4u*)(pa + 16);
      }
#pragma unroll
      for (int nt = 0; nt < 4; ++nt) {
        const unsigned short* pb = sB + (rb + nt * 16) * LDS_STRIDE + w * 32 + kh;
        bfr[nt].q[0] = *(const v4u*)(pb);
        bfr[nt].q[1] = *(const v4u*)(pb + 16);
      }
#pragma unroll
      for (int mt = 0; mt < 2; ++mt)
#pragma unroll
        for (int nt = 0; nt < 4; ++nt)
          acc[mt][nt] = __builtin_amdgcn_wmma_f32_16x16x32_bf16(
              false, af[mt].v, false, bfr[nt].v,
              (short)0, acc[mt][nt], false, false);
    }
    __syncthreads();                     // reads done before buf cur is re-filled
  }

  // C layout: VGPR j, lanes 0-15 -> D=j, lanes 16-31 -> D=j+8; N = lane&15
  float* ob = out + (size_t)b * Dc * Nc;
  const int dl = 8 * (lane >> 4);
  const int nl = lane & 15;
#pragma unroll
  for (int mt = 0; mt < 2; ++mt)
#pragma unroll
    for (int nt = 0; nt < 4; ++nt) {
      const int dbase = d0 + dw + mt * 16 + dl;
      const int nn    = n0 + nw + nt * 16 + nl;
#pragma unroll
      for (int j = 0; j < 8; ++j)
        ob[(size_t)(dbase + j) * Nc + nn] = acc[mt][nt][j];
    }
}

// ---------------------------------------------------------------------------
extern "C" void kernel_launch(void* const* d_in, const int* in_sizes, int n_in,
                              void* d_out, int out_size, void* d_ws, size_t ws_size,
                              hipStream_t stream) {
  const float* u = (const float*)d_in[0];            // [B, M, D] f32
  const float* s = (const float*)d_in[1];            // [B, N, M] f32
  float* out = (float*)d_out;                        // [B, D, N] f32

  unsigned short* wsP  = (unsigned short*)d_ws;           // bf16 [B,N,M]
  unsigned short* wsUt = wsP + (size_t)Bc * Nc * Mc;      // bf16 [B,D,M]

  softmax_rows<<<Bc * Nc, 256, 0, stream>>>(s, wsP);
  transpose_u<<<dim3(Mc / 32, Dc / 32, Bc), 256, 0, stream>>>(u, wsUt);
  c2q_gemm<<<dim3(Nc / 128, Dc / 128, Bc), 256, 0, stream>>>(wsP, wsUt, out);
}
